// LuongAttnDecoderRNN_19464791785827
// MI455X (gfx1250) — matrix operand order repaired and checked by reading, not compiled, not run on hardware
//
#include <hip/hip_runtime.h>
#include <hip/hip_bf16.h>
#include <math.h>

// ---------------------------------------------------------------------------
// Types for CDNA5 WMMA (wave32): v_wmma_f32_16x16x32_bf16
// ---------------------------------------------------------------------------
typedef __attribute__((ext_vector_type(16))) __bf16 v16bf;
typedef __attribute__((ext_vector_type(2)))  __bf16 v2bf;
typedef __attribute__((ext_vector_type(2)))  float  v2f;
typedef __attribute__((ext_vector_type(8)))  float  v8f;

// Pack two f32 -> packed bf16 pair (RNE). Vector fptrunc selects a single
// v_cvt_pk_bf16_f32 instead of two scalar converts + shift/or merging.
__device__ __forceinline__ unsigned int pack2(float a, float b) {
  v2f f = {a, b};
  union { v2bf h; unsigned int u; } cv;
  cv.h = __builtin_convertvector(f, v2bf);
  return cv.u;
}

// Async copy of 16 bytes global -> LDS (ASYNCcnt-tracked, no VGPR round trip).
// ldsOff = low 32 bits of generic LDS pointer (= in-allocation byte offset).
__device__ __forceinline__ void async_g2lds_b128(unsigned int ldsOff,
                                                 const void* gaddr) {
  asm volatile("global_load_async_to_lds_b128 %0, %1, off"
               :: "v"(ldsOff), "v"((unsigned long long)gaddr)
               : "memory");
}
__device__ __forceinline__ void async_wait0() {
  asm volatile("s_wait_asynccnt 0x0" ::: "memory");
}

#define LDSU 18  // dwords per LDS tile row (16 data + 2 pad; 72B -> no bank conflicts)

// ---------------------------------------------------------------------------
// WMMA GEMM: C[64,N] = act( A[64,K] @ W[N,K]^T + bias[N] )
// M fixed 64, K multiple of 32. Block tile 64x128, 256 threads = 8 waves.
// Wave w: rows (w&1)*32..+32, cols (w>>1)*32..+32 -> 4 wmma per K-step.
// Software-pipelined: next K-tile's b128 loads issued before fragment math.
// ---------------------------------------------------------------------------
__global__ __launch_bounds__(256)
void gemm64_wmma_bf16(const float* __restrict__ A, const float* __restrict__ W,
                      const float* __restrict__ bias, float* __restrict__ C,
                      int N, int K, int act) {
  __shared__ unsigned int As32[64 * LDSU];
  __shared__ unsigned int Ws32[128 * LDSU];
  const unsigned short* As16 = (const unsigned short*)As32;
  const unsigned short* Ws16 = (const unsigned short*)Ws32;

  const int tid   = threadIdx.x;
  const int lane  = tid & 31;
  const int wave  = tid >> 5;
  const int mh    = wave & 1;    // M half   (rows mh*32 .. +32)
  const int nq    = wave >> 1;   // N quarter (cols nq*32 .. +32)
  const int nBase = blockIdx.x * 128;

  const int m16 = lane & 15;
  const int g   = lane >> 4;

  v8f c00 = {}, c01 = {}, c10 = {}, c11 = {};

  float4 aR[2], wR[4];

  auto loadTiles = [&](int kk0) {
#pragma unroll
    for (int i = 0; i < 2; ++i) {           // A: 64 rows x 32 k = 512 float4
      int idx = tid + 256 * i;
      int r = idx >> 3, cq = (idx & 7) << 2;
      aR[i] = *(const float4*)(A + (size_t)r * K + kk0 + cq);
    }
#pragma unroll
    for (int i = 0; i < 4; ++i) {           // W: 128 rows x 32 k = 1024 float4
      int idx = tid + 256 * i;
      int r = idx >> 3, cq = (idx & 7) << 2;
      int n = nBase + r;                    // clamp tail: no divergent guard;
      if (n > N - 1) n = N - 1;             // junk rows feed only junk columns
      wR[i] = *(const float4*)(W + (size_t)n * K + kk0 + cq);
    }
  };

  loadTiles(0);  // prologue

  for (int k0 = 0; k0 < K; k0 += 32) {
    // ---- convert + store staged registers to LDS (packed bf16 pairs) ----
#pragma unroll
    for (int i = 0; i < 2; ++i) {
      int idx = tid + 256 * i;
      int r = idx >> 3, cq = (idx & 7) << 2;
      As32[r * LDSU + (cq >> 1)]     = pack2(aR[i].x, aR[i].y);
      As32[r * LDSU + (cq >> 1) + 1] = pack2(aR[i].z, aR[i].w);
    }
#pragma unroll
    for (int i = 0; i < 4; ++i) {
      int idx = tid + 256 * i;
      int r = idx >> 3, cq = (idx & 7) << 2;
      Ws32[r * LDSU + (cq >> 1)]     = pack2(wR[i].x, wR[i].y);
      Ws32[r * LDSU + (cq >> 1) + 1] = pack2(wR[i].z, wR[i].w);
    }
    __syncthreads();

    // ---- issue next tile's global loads (overlap with matrix math) ----
    if (k0 + 32 < K) loadTiles(k0 + 32);

    // ---- build fragments per documented 16-bit A(16x32)/B(32x16) layouts ----
    v16bf a0 = {}, a1 = {}, b0 = {}, b1 = {};
#pragma unroll
    for (int v = 0; v < 8; ++v) {
      union { unsigned short u; __bf16 h; } cv;
      // A: lane holds row m; VGPR v holds K pair. group0: K 0..7/16..23,
      // group1: K 8..15/24..31
      int ka = (v < 4) ? (8 * g + 2 * v) : (16 + 8 * g + 2 * (v - 4));
      int am0 = (mh * 32 + m16) * (2 * LDSU);
      int am1 = am0 + 16 * (2 * LDSU);
      cv.u = As16[am0 + ka];     a0[2 * v]     = cv.h;
      cv.u = As16[am0 + ka + 1]; a0[2 * v + 1] = cv.h;
      cv.u = As16[am1 + ka];     a1[2 * v]     = cv.h;
      cv.u = As16[am1 + ka + 1]; a1[2 * v + 1] = cv.h;
      // B: lane holds column n; same K order as A. group0: K 0..7/16..23,
      // group1: K 8..15/24..31
      int kb = (v < 4) ? (8 * g + 2 * v) : (16 + 8 * g + 2 * (v - 4));
      int bn0 = (nq * 32 + m16) * (2 * LDSU);
      int bn1 = bn0 + 16 * (2 * LDSU);
      cv.u = Ws16[bn0 + kb];     b0[2 * v]     = cv.h;
      cv.u = Ws16[bn0 + kb + 1]; b0[2 * v + 1] = cv.h;
      cv.u = Ws16[bn1 + kb];     b1[2 * v]     = cv.h;
      cv.u = Ws16[bn1 + kb + 1]; b1[2 * v + 1] = cv.h;
    }

    // 4 independent WMMAs (no D->A/B RAW between them)
    c00 = __builtin_amdgcn_wmma_f32_16x16x32_bf16(false, a0, false, b0,
                                                  (short)0, c00, false, false);
    c01 = __builtin_amdgcn_wmma_f32_16x16x32_bf16(false, a0, false, b1,
                                                  (short)0, c01, false, false);
    c10 = __builtin_amdgcn_wmma_f32_16x16x32_bf16(false, a1, false, b0,
                                                  (short)0, c10, false, false);
    c11 = __builtin_amdgcn_wmma_f32_16x16x32_bf16(false, a1, false, b1,
                                                  (short)0, c11, false, false);
    __syncthreads();
  }

  // ---- store: C/D layout: VGPR v -> row v + 8*(lane>>4), col = lane&15 ----
  int col0 = nBase + nq * 32 + m16;
  int col1 = col0 + 16;
#pragma unroll
  for (int v = 0; v < 8; ++v) {
    int r0 = mh * 32 + v + 8 * g;
    int r1 = r0 + 16;
    if (col0 < N) {
      float v0 = c00[v] + bias[col0];
      float v1 = c10[v] + bias[col0];
      if (act) { v0 = tanhf(v0); v1 = tanhf(v1); }
      C[(size_t)r0 * N + col0] = v0;
      C[(size_t)r1 * N + col0] = v1;
    }
    if (col1 < N) {
      float v0 = c01[v] + bias[col1];
      float v1 = c11[v] + bias[col1];
      if (act) { v0 = tanhf(v0); v1 = tanhf(v1); }
      C[(size_t)r0 * N + col1] = v0;
      C[(size_t)r1 * N + col1] = v1;
    }
  }
}

// ---------------------------------------------------------------------------
// Embedding gather: x[b,:] = emb[ids[b],:]
// ---------------------------------------------------------------------------
__global__ __launch_bounds__(256)
void gather_embed(const int* __restrict__ ids, const float* __restrict__ emb,
                  float* __restrict__ x) {
  int b = blockIdx.x;
  int row = ids[b];
  for (int i = threadIdx.x; i < 1024; i += 256)
    x[b * 1024 + i] = emb[(size_t)row * 1024 + i];
}

// ---------------------------------------------------------------------------
// GRU gates (PyTorch layout r,z,n): h_new = (1-z)*n + z*h
// ---------------------------------------------------------------------------
__global__ __launch_bounds__(256)
void gru_gates(const float* __restrict__ gi, const float* __restrict__ gh,
               const float* __restrict__ hPrev, float* __restrict__ hNew,
               float* __restrict__ hiddenOut) {
  int idx = blockIdx.x * 256 + threadIdx.x;  // 0..65535
  int b = idx >> 10, j = idx & 1023;
  const float* gib = gi + (size_t)b * 3072;
  const float* ghb = gh + (size_t)b * 3072;
  float ir = gib[j], iz = gib[1024 + j], in_ = gib[2048 + j];
  float hr = ghb[j], hz = ghb[1024 + j], hn = ghb[2048 + j];
  float r = 1.0f / (1.0f + expf(-(ir + hr)));
  float z = 1.0f / (1.0f + expf(-(iz + hz)));
  float n = tanhf(in_ + r * hn);
  float h = hPrev[idx];
  float out = (1.0f - z) * n + z * h;
  hNew[idx] = out;
  hiddenOut[idx] = out;
}

// ---------------------------------------------------------------------------
// Attention scores: scores[b,s] = dot(hNew[b,:], enc[b,s,:])
// hNew[b] staged to LDS via async global->LDS b128 (ASYNCcnt path).
// One wave per s-row, lanes K-strided (coalesced 128B/wave). grid (64, 8).
// ---------------------------------------------------------------------------
__global__ __launch_bounds__(256)
void attn_scores(const float* __restrict__ hNew, const float* __restrict__ enc,
                 float* __restrict__ scores) {
  __shared__ __align__(16) float hs[1024];
  int b = blockIdx.x, chunk = blockIdx.y;
  {
    unsigned int ldsOff = (unsigned int)(unsigned long long)(const void*)&hs[threadIdx.x * 4];
    async_g2lds_b128(ldsOff, hNew + b * 1024 + threadIdx.x * 4);
    async_wait0();
  }
  __syncthreads();
  int wave = threadIdx.x >> 5, lane = threadIdx.x & 31;
  const float* encB = enc + (size_t)b * 1024 * 1024;
  for (int r = 0; r < 16; ++r) {
    int s = chunk * 128 + wave * 16 + r;
    const float* row = encB + (size_t)s * 1024;
    float acc = 0.0f;
    for (int k = lane; k < 1024; k += 32) acc = fmaf(row[k], hs[k], acc);
    for (int off = 16; off; off >>= 1) acc += __shfl_xor(acc, off, 32);
    if (lane == 0) scores[b * 1024 + s] = acc;
  }
}

// ---------------------------------------------------------------------------
// Softmax over S=1024, in place. One block per batch row.
// ---------------------------------------------------------------------------
__global__ __launch_bounds__(256)
void softmax1024(float* __restrict__ attn) {
  __shared__ float red[256];
  int b = blockIdx.x;
  float* row = attn + b * 1024;
  float m = -__builtin_inff();
  for (int i = threadIdx.x; i < 1024; i += 256) m = fmaxf(m, row[i]);
  red[threadIdx.x] = m; __syncthreads();
  for (int s = 128; s; s >>= 1) {
    if (threadIdx.x < s) red[threadIdx.x] = fmaxf(red[threadIdx.x], red[threadIdx.x + s]);
    __syncthreads();
  }
  m = red[0]; __syncthreads();
  float sum = 0.0f;
  for (int i = threadIdx.x; i < 1024; i += 256) sum += expf(row[i] - m);
  red[threadIdx.x] = sum; __syncthreads();
  for (int s = 128; s; s >>= 1) {
    if (threadIdx.x < s) red[threadIdx.x] += red[threadIdx.x + s];
    __syncthreads();
  }
  float inv = 1.0f / red[0];
  __syncthreads();
  for (int i = threadIdx.x; i < 1024; i += 256) row[i] = expf(row[i] - m) * inv;
}

// ---------------------------------------------------------------------------
// context[b,h] = sum_s attn[b,s]*enc[b,s,h]; also assemble concatIn = [h|ctx]
// attn[b] staged to LDS via async global->LDS b128 (ASYNCcnt path).
// grid (64, 4), 256 threads -> thread owns one h column (coalesced over h).
// ---------------------------------------------------------------------------
__global__ __launch_bounds__(256)
void context_kernel(const float* __restrict__ attn, const float* __restrict__ enc,
                    const float* __restrict__ hNew, float* __restrict__ concatIn) {
  __shared__ __align__(16) float aw[1024];
  int b = blockIdx.x;
  int h = blockIdx.y * 256 + threadIdx.x;
  {
    unsigned int ldsOff = (unsigned int)(unsigned long long)(const void*)&aw[threadIdx.x * 4];
    async_g2lds_b128(ldsOff, attn + b * 1024 + threadIdx.x * 4);
    async_wait0();
  }
  __syncthreads();
  const float* encB = enc + (size_t)b * 1024 * 1024;
  float acc = 0.0f;
  for (int s = 0; s < 1024; ++s) acc = fmaf(aw[s], encB[(size_t)s * 1024 + h], acc);
  concatIn[(size_t)b * 2048 + h] = hNew[b * 1024 + h];
  concatIn[(size_t)b * 2048 + 1024 + h] = acc;
}

// ---------------------------------------------------------------------------
// Row log-sum-exp over V=50257 (one block per batch row)
// ---------------------------------------------------------------------------
__global__ __launch_bounds__(256)
void row_lse(const float* __restrict__ logits, float* __restrict__ lse) {
  __shared__ float red[256];
  int b = blockIdx.x;
  const float* row = logits + (size_t)b * 50257;
  float m = -__builtin_inff();
  for (int i = threadIdx.x; i < 50257; i += 256) m = fmaxf(m, row[i]);
  red[threadIdx.x] = m; __syncthreads();
  for (int s = 128; s; s >>= 1) {
    if (threadIdx.x < s) red[threadIdx.x] = fmaxf(red[threadIdx.x], red[threadIdx.x + s]);
    __syncthreads();
  }
  m = red[0]; __syncthreads();
  float sum = 0.0f;
  for (int i = threadIdx.x; i < 50257; i += 256) sum += expf(row[i] - m);
  red[threadIdx.x] = sum; __syncthreads();
  for (int s = 128; s; s >>= 1) {
    if (threadIdx.x < s) red[threadIdx.x] += red[threadIdx.x + s];
    __syncthreads();
  }
  if (threadIdx.x == 0) lse[b] = m + logf(red[0]);
}

__global__ __launch_bounds__(256)
void sub_lse(float* __restrict__ out, const float* __restrict__ lse) {
  size_t idx = (size_t)blockIdx.x * 256 + threadIdx.x;
  if (idx >= (size_t)64 * 50257) return;
  int b = (int)(idx / 50257);
  out[idx] -= lse[b];
}

// ---------------------------------------------------------------------------
// Launch
// ---------------------------------------------------------------------------
extern "C" void kernel_launch(void* const* d_in, const int* in_sizes, int n_in,
                              void* d_out, int out_size, void* d_ws, size_t ws_size,
                              hipStream_t stream) {
  (void)in_sizes; (void)n_in; (void)out_size; (void)ws_size;

  const int*   ids   = (const int*)d_in[0];    // [64]
  const float* hPrev = (const float*)d_in[1];  // [1,64,1024]
  const float* enc   = (const float*)d_in[2];  // [64,1024,1024]
  const float* emb   = (const float*)d_in[3];  // [50257,1024]
  const float* w_ih  = (const float*)d_in[4];  // [3072,1024]
  const float* w_hh  = (const float*)d_in[5];  // [3072,1024]
  const float* b_ih  = (const float*)d_in[6];  // [3072]
  const float* b_hh  = (const float*)d_in[7];  // [3072]
  const float* c_w   = (const float*)d_in[8];  // [1024,2048]
  const float* c_b   = (const float*)d_in[9];  // [1024]
  const float* o_w   = (const float*)d_in[10]; // [50257,1024]
  const float* o_b   = (const float*)d_in[11]; // [50257]

  float* out = (float*)d_out;  // [64*50257] log-probs, then [64*1024] hidden
  float* logits    = out;
  float* hiddenOut = out + (size_t)64 * 50257;

  float* ws = (float*)d_ws;
  float* x         = ws;                   // 65536
  float* gi        = x + 65536;            // 196608
  float* gh        = gi + 196608;          // 196608
  float* hNew      = gh + 196608;          // 65536
  float* attn      = hNew + 65536;         // 65536
  float* concatIn  = attn + 65536;         // 131072
  float* concatOut = concatIn + 131072;    // 65536
  float* lse       = concatOut + 65536;    // 64

  gather_embed<<<64, 256, 0, stream>>>(ids, emb, x);
  gemm64_wmma_bf16<<<24, 256, 0, stream>>>(x, w_ih, b_ih, gi, 3072, 1024, 0);
  gemm64_wmma_bf16<<<24, 256, 0, stream>>>(hPrev, w_hh, b_hh, gh, 3072, 1024, 0);
  gru_gates<<<256, 256, 0, stream>>>(gi, gh, hPrev, hNew, hiddenOut);
  attn_scores<<<dim3(64, 8), 256, 0, stream>>>(hNew, enc, attn);
  softmax1024<<<64, 256, 0, stream>>>(attn);
  context_kernel<<<dim3(64, 4), 256, 0, stream>>>(attn, enc, hNew, concatIn);
  gemm64_wmma_bf16<<<8, 256, 0, stream>>>(concatIn, c_w, c_b, concatOut, 1024, 2048, 1);
  gemm64_wmma_bf16<<<(50257 + 127) / 128, 256, 0, stream>>>(concatOut, o_w, o_b,
                                                            logits, 50257, 1024, 0);
  row_lse<<<64, 256, 0, stream>>>(logits, lse);
  sub_lse<<<(64 * 50257 + 255) / 256, 256, 0, stream>>>(logits, lse);
}
